// ConditionalDLFactorized15_74680891343525
// MI455X (gfx1250) — matrix operand order repaired and checked by reading, not compiled, not run on hardware
//
#include <hip/hip_runtime.h>
#include <math.h>

// ---------------------------------------------------------------------------
// ConditionalDLFactorized: fully algebraically-refactored implementation.
//   keyf / quantized / w2 (3 x 512MB intermediates) are never materialized.
//   Heavy GEMMs run on v_wmma_f32_16x16x32_bf16 (f32 accumulate).
//   NN-shaped GEMM (P / P2) stages W tiles into LDS via the Tensor Data
//   Mover (tensor_load_to_lds, double-buffered, s_wait_tensorcnt).
// ---------------------------------------------------------------------------

namespace cfg {
constexpr int T_ = 512, B_ = 4, C_ = 256;
constexpr int N_ = T_ * B_;          // 2048 tokens
constexpr int NE_ = 64;              // experts
constexpr int D_ = 65536;            // R*OUT
constexpr int OUT_ = 256;
constexpr float DECAY_ = 0.999f;
constexpr float ALPHA_ = (1.0f - DECAY_) / (float)N_;
constexpr float COMMIT_ = 0.25f;
}

typedef __attribute__((ext_vector_type(16))) __bf16 bf16x16;
typedef __attribute__((ext_vector_type(8)))  float  floatx8;
typedef __attribute__((ext_vector_type(4)))  unsigned int uint32x4;
typedef __attribute__((ext_vector_type(8)))  int int32x8;
typedef __attribute__((ext_vector_type(4)))  int int32x4;

// ---------------- WMMA fragment helpers (wave32; A/B k contiguous, not ISA table order) -------------
// A (16x32 bf16, MxK): lane l<16 -> row M=l, elems 0..15 = K0..15
//                      lane l>=16 -> row M=l-16, K16..31
__device__ __forceinline__ bf16x16 frag_a_rm(const float* __restrict__ A,
                                             int lda, int row, int k0, int lane) {
  const float* p = A + (size_t)row * lda + k0 + (lane >> 4) * 16;
  bf16x16 f;
#pragma unroll
  for (int i = 0; i < 8; ++i) { f[i] = (__bf16)p[i]; f[i + 8] = (__bf16)p[i + 8]; }
  return f;
}
// B (32x16 bf16, KxN), operand stored row-major [Ncol][K] (NT gemm):
// lane l<16 -> col N=l, elems 0..15 = K0..15; lane>=16 -> K16..31
__device__ __forceinline__ bf16x16 frag_b_rm(const float* __restrict__ B,
                                             int ldb, int col, int k0, int lane) {
  const float* p = B + (size_t)col * ldb + k0 + (lane >> 4) * 16;
  bf16x16 f;
#pragma unroll
  for (int i = 0; i < 16; ++i) f[i] = (__bf16)p[i];
  return f;
}
__device__ __forceinline__ floatx8 wmma_bf16(bf16x16 a, bf16x16 b, floatx8 c) {
  return __builtin_amdgcn_wmma_f32_16x16x32_bf16(false, a, false, b,
                                                 (short)0, c, false, false);
}
// D (16x16 f32): vgpr r -> row r (lanes 0-15) / r+8 (lanes 16-31); col = lane&15
__device__ __forceinline__ void store_d_tile(float* __restrict__ O, int ldo,
                                             int rbase, int cbase, int lane,
                                             floatx8 acc) {
  int col = cbase + (lane & 15);
  int roff = (lane >> 4) * 8;
#pragma unroll
  for (int r = 0; r < 8; ++r) O[(size_t)(rbase + r + roff) * ldo + col] = acc[r];
}

__device__ __forceinline__ void block_reduce_atomic256(float v, float* tgt) {
  __shared__ float sm[256];
  sm[threadIdx.x] = v;
  __syncthreads();
  for (int s = 128; s > 0; s >>= 1) {
    if ((int)threadIdx.x < s) sm[threadIdx.x] += sm[threadIdx.x + s];
    __syncthreads();
  }
  if (threadIdx.x == 0) atomicAdd(tgt, sm[0]);
}

// ---- Tensor Data Mover: DMA W[k0:k0+64, j0:j0+32] (f32) into LDS ----------
// D# per CDNA5 ISA: group0 = {count=1, lds_addr,
// global_addr, type=2}; group1 = {data_size=4B, tensor/tile dims, stride}.
// This toolchain exposes the 6-arg builtin form (extra int32x8 group, zeroed).
__device__ __forceinline__ void tdm_load_w_tile(const float* gptr,
                                                unsigned lds_byte_off) {
  unsigned long long ga = (unsigned long long)(size_t)gptr;
  uint32x4 g0;
  g0[0] = 1u;                                        // count=1, load, no gather
  g0[1] = lds_byte_off;                              // LDS destination
  g0[2] = (unsigned int)(ga & 0xFFFFFFFFu);          // global_addr[31:0]
  g0[3] = (unsigned int)((ga >> 32) & 0x01FFFFFFu)   // global_addr[56:32]
          | 0x80000000u;                             // type=2 (bits 127:126)
  int32x8 g1;
  g1[0] = 2 << 16;        // workgroup_mask=0, data_size=2 (4 bytes)
  g1[1] = 32 << 16;       // tensor_dim0 = 32 (tile-local x extent)
  g1[2] = 64 << 16;       // tensor_dim0 hi | tensor_dim1 = 64
  g1[3] = 32 << 16;       // tensor_dim1 hi | tile_dim0 = 32
  g1[4] = 64;             // tile_dim1 = 64, tile_dim2 = 0
  g1[5] = 256;            // tensor_dim0_stride = 256 (row pitch of W)
  g1[6] = 0;
  g1[7] = 0;
  int32x4 gz = {0, 0, 0, 0};
  int32x8 gz8 = {0, 0, 0, 0, 0, 0, 0, 0};
  __builtin_amdgcn_tensor_load_to_lds(g0, g1, gz, gz, gz8, 0);
}

using namespace cfg;

// ---------------------------------------------------------------------------
__global__ __launch_bounds__(64) void k_init(float* sE, float* scal) {
  int t = threadIdx.x;
  if (t < NE_) sE[t] = 0.f;
  if (t < 4) scal[t] = 0.f;
}

// cnorm[e] = ||cent_e||^2 ; bc[e] = b . cent_e
__global__ __launch_bounds__(256) void k_cnorm(const float* __restrict__ cent,
                                               const float* __restrict__ b,
                                               float* cnorm, float* bc) {
  int e = blockIdx.x, t = threadIdx.x;
  float a2 = 0.f, ab = 0.f;
  for (int d = t; d < D_; d += 256) {
    float v = cent[(size_t)e * D_ + d];
    a2 += v * v;
    ab += b[d] * v;
  }
  __shared__ float s1[256], s2[256];
  s1[t] = a2; s2[t] = ab;
  __syncthreads();
  for (int s = 128; s > 0; s >>= 1) {
    if (t < s) { s1[t] += s1[t + s]; s2[t] += s2[t + s]; }
    __syncthreads();
  }
  if (t == 0) { cnorm[e] = s1[0]; bc[e] = s2[0]; }
}

// Out[e,j] = sum_d Arows[e,d] * W[d,j]   (M=64, N=256, K=65536)  [P and P2]
// Block owns a 32-wide j-slab; 8 waves = 4 m-tiles x 2 n-tiles.
// W K-slabs are DMA'd into LDS by the TDM, double-buffered so transfer of
// slab i+1 overlaps WMMA on slab i.
__global__ __launch_bounds__(256) void k_gemm_nn(const float* __restrict__ Arows,
                                                 const float* __restrict__ W,
                                                 float* __restrict__ Out) {
  __shared__ float tile[2][64 * 32];   // [buf][k-local * 32 + j-local]
  int lane = threadIdx.x & 31;
  int wave = threadIdx.x >> 5;
  int j0 = blockIdx.x * 32;
  int tm = wave >> 1;                  // e tile 0..3
  int jt = (wave & 1) * 16;            // local j tile
  int row = tm * 16 + (lane & 15);
  int jl = jt + (lane & 15);
  floatx8 acc = {0.f, 0.f, 0.f, 0.f, 0.f, 0.f, 0.f, 0.f};

  if (wave == 0) tdm_load_w_tile(W + (size_t)0 * C_ + j0, 0u);

  for (int k0 = 0; k0 < D_; k0 += 64) {
    int ib = (k0 >> 6) & 1;
    if (wave == 0) __builtin_amdgcn_s_wait_tensorcnt(0);
    __syncthreads();                   // tile[ib] ready for all waves
    if (wave == 0 && (k0 + 64) < D_)
      tdm_load_w_tile(W + (size_t)(k0 + 64) * C_ + j0,
                      (unsigned)((ib ^ 1) * 64 * 32 * 4));
#pragma unroll
    for (int ks = 0; ks < 64; ks += 32) {
      bf16x16 a = frag_a_rm(Arows, D_, row, k0 + ks, lane);
      int kb = ks + (lane >> 4) * 16;
      bf16x16 b;
#pragma unroll
      for (int i = 0; i < 16; ++i) b[i] = (__bf16)tile[ib][(kb + i) * 32 + jl];
      acc = wmma_bf16(a, b, acc);
    }
    __syncthreads();                   // done reading tile[ib]
  }
  store_d_tile(Out, C_, tm * 16, j0 + jt, lane, acc);
}

// routing: logits (||kd||^2 term cancels in softmax), gumbel, softmax, resp
__global__ __launch_bounds__(64) void k_resp(const float* __restrict__ x,
                                             const float* __restrict__ un,
                                             const float* __restrict__ P,
                                             const float* __restrict__ cnorm,
                                             const float* __restrict__ bc,
                                             float* __restrict__ resp,
                                             float* __restrict__ sE) {
  int n = blockIdx.x, e = threadIdx.x;
  const float* xr = x + (size_t)n * C_;
  const float* pr = P + (size_t)e * C_;
  float s0 = bc[e];
  for (int j = 0; j < C_; ++j) s0 += xr[j] * pr[j];
  float u = un[(size_t)n * NE_ + e];
  float g = -__logf(-__logf(u));
  float logit = 2.f * s0 - cnorm[e] + g;  // TAU == 1
  __shared__ float sm[64];
  sm[e] = logit;
  __syncthreads();
  for (int s = 32; s > 0; s >>= 1) {
    if (e < s) sm[e] = fmaxf(sm[e], sm[e + s]);
    __syncthreads();
  }
  float mx = sm[0];
  __syncthreads();
  float ex = __expf(logit - mx);
  sm[e] = ex;
  __syncthreads();
  for (int s = 32; s > 0; s >>= 1) {
    if (e < s) sm[e] += sm[e + s];
    __syncthreads();
  }
  float r = ex / sm[0];
  resp[(size_t)n * NE_ + e] = r;
  atomicAdd(&sE[e], r);
}

// A[e,j] = sum_n resp[n,e] x[n,j]  (exact f32, tiny)
__global__ __launch_bounds__(256) void k_A(const float* __restrict__ resp,
                                           const float* __restrict__ x,
                                           float* __restrict__ A) {
  int e = blockIdx.x, j = threadIdx.x;
  float acc = 0.f;
  for (int n = 0; n < N_; ++n)
    acc += resp[(size_t)n * NE_ + e] * x[(size_t)n * C_ + j];
  A[(size_t)e * C_ + j] = acc;
}

// C_new[e,d] = DECAY*cent + ALPHA*(sum_j A[e,j] W[d,j] + s[e]*b[d])
__global__ __launch_bounds__(256) void k_cnew(const float* __restrict__ A,
                                              const float* __restrict__ W,
                                              const float* __restrict__ cent,
                                              const float* __restrict__ b,
                                              const float* __restrict__ sE,
                                              float* __restrict__ Cn) {
  int gw = (blockIdx.x * blockDim.x + threadIdx.x) >> 5;
  int lane = threadIdx.x & 31;
  int tm = gw / (D_ / 16), tn = gw % (D_ / 16);
  int row = tm * 16 + (lane & 15);  // e
  int col = tn * 16 + (lane & 15);  // d
  floatx8 acc = {0.f, 0.f, 0.f, 0.f, 0.f, 0.f, 0.f, 0.f};
  for (int k0 = 0; k0 < C_; k0 += 32) {
    bf16x16 a = frag_a_rm(A, C_, row, k0, lane);
    bf16x16 bb = frag_b_rm(W, C_, col, k0, lane);
    acc = wmma_bf16(a, bb, acc);
  }
  int roff = (lane >> 4) * 8;
#pragma unroll
  for (int r = 0; r < 8; ++r) {
    int e_ = tm * 16 + r + roff;
    size_t idx = (size_t)e_ * D_ + col;
    Cn[idx] = DECAY_ * cent[idx] + ALPHA_ * (acc[r] + sE[e_] * b[col]);
  }
}

// bc2[e] = b . C_new[e]
__global__ __launch_bounds__(256) void k_bc2(const float* __restrict__ b,
                                             const float* __restrict__ Cn,
                                             float* __restrict__ bc2) {
  int e = blockIdx.x, t = threadIdx.x;
  float ab = 0.f;
  for (int d = t; d < D_; d += 256) ab += b[d] * Cn[(size_t)e * D_ + d];
  __shared__ float s1[256];
  s1[t] = ab;
  __syncthreads();
  for (int s = 128; s > 0; s >>= 1) {
    if (t < s) s1[t] += s1[t + s];
    __syncthreads();
  }
  if (t == 0) bc2[e] = s1[0];
}

// cross += sum_e resp[n,e]*(x[n].P2[e] + bc2[e])
__global__ __launch_bounds__(64) void k_cross(const float* __restrict__ x,
                                              const float* __restrict__ P2,
                                              const float* __restrict__ bc2,
                                              const float* __restrict__ resp,
                                              float* __restrict__ crossAcc) {
  int n = blockIdx.x, e = threadIdx.x;
  const float* xr = x + (size_t)n * C_;
  const float* pr = P2 + (size_t)e * C_;
  float t = bc2[e];
  for (int j = 0; j < C_; ++j) t += xr[j] * pr[j];
  float v = resp[(size_t)n * NE_ + e] * t;
  __shared__ float sm[64];
  sm[e] = v;
  __syncthreads();
  for (int s = 32; s > 0; s >>= 1) {
    if (e < s) sm[e] += sm[e + s];
    __syncthreads();
  }
  if (e == 0) atomicAdd(crossAcc, sm[0]);
}

// G = X^T X  (256x256, symmetric)
__global__ __launch_bounds__(256) void k_G(const float* __restrict__ x,
                                           float* __restrict__ G) {
  int idx = blockIdx.x * 256 + threadIdx.x;
  int j = idx >> 8, j2 = idx & 255;
  float acc = 0.f;
  for (int n = 0; n < N_; ++n)
    acc += x[(size_t)n * C_ + j] * x[(size_t)n * C_ + j2];
  G[idx] = acc;
}

// u[j] = sum_n x[n,j]
__global__ __launch_bounds__(256) void k_u(const float* __restrict__ x,
                                           float* __restrict__ uC) {
  int j = threadIdx.x;
  float acc = 0.f;
  for (int n = 0; n < N_; ++n) acc += x[(size_t)n * C_ + j];
  uC[j] = acc;
}

// sum_d w_d^T G w_d : V = W*G (WMMA, G symmetric) fused with <V,W> epilogue
__global__ __launch_bounds__(256) void k_sumk2(const float* __restrict__ W,
                                               const float* __restrict__ G,
                                               float* __restrict__ slot) {
  int gw = (blockIdx.x * blockDim.x + threadIdx.x) >> 5;
  int lane = threadIdx.x & 31;
  int tm = gw / 16, tn = gw % 16;
  int row = tm * 16 + (lane & 15);  // d
  int col = tn * 16 + (lane & 15);  // j'
  floatx8 acc = {0.f, 0.f, 0.f, 0.f, 0.f, 0.f, 0.f, 0.f};
  for (int k0 = 0; k0 < C_; k0 += 32) {
    bf16x16 a = frag_a_rm(W, C_, row, k0, lane);
    bf16x16 bb = frag_b_rm(G, C_, col, k0, lane);  // G == G^T
    acc = wmma_bf16(a, bb, acc);
  }
  int roff = (lane >> 4) * 8;
  float p = 0.f;
#pragma unroll
  for (int r = 0; r < 8; ++r) {
    int d_ = tm * 16 + r + roff;
    p += acc[r] * W[(size_t)d_ * C_ + col];
  }
  block_reduce_atomic256(p, slot);
}

// b-terms of sum(keyf^2): sum_d 2 b_d (u.W_d) + N b_d^2
__global__ __launch_bounds__(256) void k_bterm(const float* __restrict__ W,
                                               const float* __restrict__ b,
                                               const float* __restrict__ uC,
                                               float* __restrict__ slot) {
  int d = blockIdx.x * 256 + threadIdx.x;
  float bd = b[d];
  float dot = 0.f;
  for (int j = 0; j < C_; ++j) dot += uC[j] * W[(size_t)d * C_ + j];
  float v = 2.f * bd * dot + (float)N_ * bd * bd;
  block_reduce_atomic256(v, slot);
}

// K2 = C_new C_new^T  (64x64, K=65536, WMMA NT)
__global__ __launch_bounds__(256) void k_K2(const float* __restrict__ Cn,
                                            float* __restrict__ K2) {
  int gw = (blockIdx.x * blockDim.x + threadIdx.x) >> 5;
  int lane = threadIdx.x & 31;
  if (gw >= 16) return;
  int tm = gw / 4, tn = gw % 4;
  int row = tm * 16 + (lane & 15);
  int col = tn * 16 + (lane & 15);
  floatx8 acc = {0.f, 0.f, 0.f, 0.f, 0.f, 0.f, 0.f, 0.f};
  for (int k0 = 0; k0 < D_; k0 += 32) {
    bf16x16 a = frag_a_rm(Cn, D_, row, k0, lane);
    bf16x16 bb = frag_b_rm(Cn, D_, col, k0, lane);
    acc = wmma_bf16(a, bb, acc);
  }
  store_d_tile(K2, NE_, tm * 16, tn * 16, lane, acc);
}

// sumQ2 += (sum_n resp[n,e] resp[n,e']) * K2[e,e']
__global__ __launch_bounds__(256) void k_sumq2(const float* __restrict__ resp,
                                               const float* __restrict__ K2,
                                               float* __restrict__ slot) {
  int idx = blockIdx.x * 256 + threadIdx.x;
  int e = idx >> 6, e2 = idx & 63;
  float a = 0.f;
  for (int n = 0; n < N_; ++n)
    a += resp[(size_t)n * NE_ + e] * resp[(size_t)n * NE_ + e2];
  block_reduce_atomic256(a * K2[idx], slot);
}

// Y[n,k] = sum_j pw1[k,j] x[n,j]   (M=2048, N=256, K=256, WMMA NT)
__global__ __launch_bounds__(256) void k_Y(const float* __restrict__ x,
                                           const float* __restrict__ pw1,
                                           float* __restrict__ Y) {
  int gw = (blockIdx.x * blockDim.x + threadIdx.x) >> 5;
  int lane = threadIdx.x & 31;
  int tm = gw / 16, tn = gw % 16;
  int row = tm * 16 + (lane & 15);  // n
  int col = tn * 16 + (lane & 15);  // k
  floatx8 acc = {0.f, 0.f, 0.f, 0.f, 0.f, 0.f, 0.f, 0.f};
  for (int k0 = 0; k0 < C_; k0 += 32) {
    bf16x16 a = frag_a_rm(x, C_, row, k0, lane);
    bf16x16 bb = frag_b_rm(pw1, C_, col, k0, lane);
    acc = wmma_bf16(a, bb, acc);
  }
  store_d_tile(Y, C_, tm * 16, tn * 16, lane, acc);
}

// out[n,i] = sum_{e,k} (resp[n,e]*Y[n,k]) * C_new[e, i*256+k] + pw_B[i]
// M=2048, N=256, K=16384; A operand formed on the fly (never materialized).
__global__ __launch_bounds__(256) void k_out(const float* __restrict__ resp,
                                             const float* __restrict__ Y,
                                             const float* __restrict__ Cn,
                                             const float* __restrict__ pwB,
                                             float* __restrict__ out) {
  int gw = (blockIdx.x * blockDim.x + threadIdx.x) >> 5;
  int lane = threadIdx.x & 31;
  int tm = gw / 16, tn = gw % 16;
  int hi = lane >> 4;
  int n_row = tm * 16 + (lane & 15);
  int i_col = tn * 16 + (lane & 15);
  floatx8 acc = {0.f, 0.f, 0.f, 0.f, 0.f, 0.f, 0.f, 0.f};
  for (int e = 0; e < NE_; ++e) {
    float rv = resp[(size_t)n_row * NE_ + e];
    const float* yrow = Y + (size_t)n_row * C_;
    const float* brow = Cn + (size_t)e * D_ + (size_t)i_col * C_;
    for (int kb = 0; kb < C_; kb += 32) {
      const float* yp = yrow + kb + hi * 16;
      bf16x16 a;
#pragma unroll
      for (int i = 0; i < 8; ++i) {
        a[i] = (__bf16)(rv * yp[i]);
        a[i + 8] = (__bf16)(rv * yp[i + 8]);
      }
      const float* bp = brow + kb + hi * 16;
      bf16x16 bb;
#pragma unroll
      for (int i = 0; i < 16; ++i) bb[i] = (__bf16)bp[i];
      acc = wmma_bf16(a, bb, acc);
    }
  }
  int roff = hi * 8;
  float bias = pwB[i_col];
#pragma unroll
  for (int r = 0; r < 8; ++r)
    out[(size_t)(tm * 16 + r + roff) * OUT_ + i_col] = acc[r] + bias;
}

__global__ void k_finalize(const float* __restrict__ scal,
                           float* __restrict__ out) {
  // loss = COMMIT * mean((keyf - quantized)^2)
  //      = COMMIT/(N*D) * (sum keyf^2 + bterms - 2*cross + sum q^2)
  float v = scal[0] + scal[1] - 2.f * scal[2] + scal[3];
  out[(size_t)N_ * OUT_] = COMMIT_ * v / ((float)N_ * (float)D_);
}

// ---------------------------------------------------------------------------
extern "C" void kernel_launch(void* const* d_in, const int* in_sizes, int n_in,
                              void* d_out, int out_size, void* d_ws, size_t ws_size,
                              hipStream_t stream) {
  const float* x    = (const float*)d_in[0];  // [T,B,C] = [N,C]
  const float* un   = (const float*)d_in[1];  // [N,NE]
  const float* W    = (const float*)d_in[2];  // [D,C]
  const float* b    = (const float*)d_in[3];  // [D]
  const float* pw1  = (const float*)d_in[4];  // [1,1,R,C]
  const float* pwB  = (const float*)d_in[5];  // [1,1,OUT]
  const float* cent = (const float*)d_in[6];  // [NE,D]
  float* out = (float*)d_out;                 // [N*OUT] + 1 loss scalar

  float* w = (float*)d_ws;
  float* P    = w; w += NE_ * C_;
  float* P2   = w; w += NE_ * C_;
  float* cn   = w; w += NE_;
  float* bc   = w; w += NE_;
  float* bc2  = w; w += NE_;
  float* sE   = w; w += NE_;
  float* scal = w; w += 4;              // [sumK2_main, sumK2_b, cross, sumQ2]
  float* uC   = w; w += C_;
  float* G    = w; w += C_ * C_;
  float* K2   = w; w += NE_ * NE_;
  float* resp = w; w += (size_t)N_ * NE_;
  float* A    = w; w += NE_ * C_;
  float* Y    = w; w += (size_t)N_ * C_;
  float* Cn   = w; w += (size_t)NE_ * D_;   // ~20 MB total

  k_init<<<1, 64, 0, stream>>>(sE, scal);
  k_cnorm<<<NE_, 256, 0, stream>>>(cent, b, cn, bc);
  // P = centroids @ W  (TDM-staged, 8 j-slab blocks)
  k_gemm_nn<<<8, 256, 0, stream>>>(cent, W, P);
  k_resp<<<N_, 64, 0, stream>>>(x, un, P, cn, bc, resp, sE);
  k_A<<<NE_, 256, 0, stream>>>(resp, x, A);
  // C_new: 64x65536, 16384 wave-tiles
  k_cnew<<<2048, 256, 0, stream>>>(A, W, cent, b, sE, Cn);
  k_bc2<<<NE_, 256, 0, stream>>>(b, Cn, bc2);
  // P2 = C_new @ W  (TDM-staged)
  k_gemm_nn<<<8, 256, 0, stream>>>(Cn, W, P2);
  k_cross<<<N_, 64, 0, stream>>>(x, P2, bc2, resp, &scal[2]);
  k_G<<<256, 256, 0, stream>>>(x, G);
  k_u<<<1, 256, 0, stream>>>(x, uC);
  // sum keyf^2 main term: 65536 wave-tiles
  k_sumk2<<<8192, 256, 0, stream>>>(W, G, &scal[0]);
  k_bterm<<<256, 256, 0, stream>>>(W, b, uC, &scal[1]);
  k_K2<<<2, 256, 0, stream>>>(Cn, K2);
  k_sumq2<<<16, 256, 0, stream>>>(resp, K2, &scal[3]);
  // Y = X @ pw1^T : 2048 wave-tiles
  k_Y<<<256, 256, 0, stream>>>(x, pw1, Y);
  // out GEMM (dominant, K=16384): 2048 wave-tiles
  k_out<<<256, 256, 0, stream>>>(resp, Y, Cn, pwB, out);
  k_finalize<<<1, 1, 0, stream>>>(scal, out);
}
